// FP8Experts_27419071218011
// MI455X (gfx1250) — 8-bit matrix operand order repaired and checked by reading, compiled, not run on hardware
//
#include <hip/hip_runtime.h>


typedef __attribute__((ext_vector_type(16))) int   v16i;
typedef __attribute__((ext_vector_type(8)))  float v8f;

#ifndef __has_builtin
#define __has_builtin(x) 0
#endif

#if __has_builtin(__builtin_amdgcn_global_load_async_to_lds_b128)
#define ASYNC_LDS 1
#else
#define ASYNC_LDS 0
#endif

#if ASYNC_LDS
typedef int async_v4i __attribute__((vector_size(16)));
typedef async_v4i __attribute__((address_space(1)))* async_gp;
typedef async_v4i __attribute__((address_space(3)))* async_lp;
#endif

__device__ __forceinline__ void async_cp16(const void* g, void* l) {
#if ASYNC_LDS
  __builtin_amdgcn_global_load_async_to_lds_b128(
      (async_gp)(size_t)g, (async_lp)(unsigned)(size_t)l, 0, 0);
#else
  *(uint4*)l = *(const uint4*)g;
#endif
}

#if ASYNC_LDS
# if __has_builtin(__builtin_amdgcn_s_wait_asynccnt)
#  define ASYNC_WAIT(n) __builtin_amdgcn_s_wait_asynccnt(n)
# else
#  define ASYNC_WAIT(n) asm volatile("s_wait_asynccnt %0" :: "i"(n) : "memory")
# endif
#else
# define ASYNC_WAIT(n)
#endif

#define T_TOK   8192
#define H_DIM   2048
#define I_DIM   1024
#define E_NUM   32
#define TOPK    8
#define NPAIR   (T_TOK*TOPK)
#define KB_H    (H_DIM/128)
#define KB_I    (I_DIM/128)
#define MAXTILE 544

#define OFF_COUNTS 0ULL
#define OFF_FILL   256ULL
#define OFF_OFFS   512ULL
#define OFF_NT     768ULL
#define OFF_TE     1024ULL
#define OFF_TM     4096ULL
#define OFF_XS     8192ULL
#define OFF_XQ     532480ULL
#define OFF_ROWS   17309696ULL
#define OFF_ROWW   17571840ULL
#define OFF_HS     17833984ULL
#define OFF_HQ     19931136ULL
#define OFF_W8GU   87040000ULL
#define OFF_W8D    221257728ULL

#define GU_ELEMS   (134217728LL)
#define D_ELEMS    (67108864LL)

__device__ __forceinline__ unsigned int f32_to_e4m3(float f) {
  unsigned int u  = __float_as_uint(f);
  unsigned int s  = (u >> 24) & 0x80u;
  unsigned int au = u & 0x7fffffffu;
  if (au >= 0x7f800000u) return s | 0x7eu;
  float a = __uint_as_float(au);
  if (a < 0.015625f) {
    int q = (int)rintf(a * 512.0f);
    return s | (unsigned int)q;
  }
  unsigned int mant = au & 0x7fffffu;
  unsigned int e32  = au >> 23;
  unsigned int lsb  = (mant >> 20) & 1u;
  mant += 0x7ffffu + lsb;
  if (mant & 0x800000u) { e32 += 1; mant &= 0x7fffffu; }
  int e8 = (int)e32 - 120;
  unsigned int m3 = mant >> 20;
  if (e8 > 15 || (e8 == 15 && m3 > 6u)) return s | 0x7eu;
  return s | ((unsigned int)e8 << 3) | m3;
}

__device__ __forceinline__ unsigned int pack4_e4m3(float4 v) {
  return  f32_to_e4m3(v.x)        | (f32_to_e4m3(v.y) << 8) |
         (f32_to_e4m3(v.z) << 16) | (f32_to_e4m3(v.w) << 24);
}

__global__ void k_zero_small(int* counts, int* fill) {
  int i = threadIdx.x;
  if (i < E_NUM) { counts[i] = 0; fill[i] = 0; }
}

__global__ void k_zero4(float* p, long long n4) {
  long long i = (long long)blockIdx.x * blockDim.x + threadIdx.x;
  if (i < n4) ((float4*)p)[i] = make_float4(0.f, 0.f, 0.f, 0.f);
}

__global__ void k_count(const int* __restrict__ idx, int* __restrict__ counts) {
  int p = blockIdx.x * blockDim.x + threadIdx.x;
  if (p < NPAIR) atomicAdd(&counts[idx[p]], 1);
}

__global__ void k_scan(const int* __restrict__ counts, int* __restrict__ offs,
                       int* __restrict__ te, int* __restrict__ tm,
                       int* __restrict__ nt) {
  if (threadIdx.x == 0) {
    int acc = 0, n = 0;
    for (int e = 0; e < E_NUM; ++e) {
      offs[e] = acc;
      int c = counts[e];
      int t = (c + 127) >> 7;
      for (int i = 0; i < t; ++i) { te[n] = e; tm[n] = i; ++n; }
      acc += c;
    }
    offs[E_NUM] = acc;
    nt[0] = n;
  }
}

__global__ void k_fill(const int* __restrict__ idx, const float* __restrict__ tkw,
                       const int* __restrict__ offs, int* __restrict__ fill,
                       int* __restrict__ rows, float* __restrict__ roww) {
  int p = blockIdx.x * blockDim.x + threadIdx.x;
  if (p < NPAIR) {
    int e   = idx[p];
    int pos = atomicAdd(&fill[e], 1);
    int slot = offs[e] + pos;
    rows[slot] = p >> 3;
    roww[slot] = tkw[p];
  }
}

__global__ __launch_bounds__(128)
void k_actq(const float* __restrict__ x, unsigned char* __restrict__ xq,
            float* __restrict__ xs) {
  int b  = blockIdx.x;
  int t  = b >> 4;
  int kb = b & 15;
  int i  = threadIdx.x;
  size_t off = (size_t)t * H_DIM + (size_t)kb * 128 + i;
  float a = x[off];
  float m = fabsf(a);
  #pragma unroll
  for (int o = 16; o > 0; o >>= 1) m = fmaxf(m, __shfl_xor(m, o, 32));
  __shared__ float red[4];
  if ((i & 31) == 0) red[i >> 5] = m;
  __syncthreads();
  float amax  = fmaxf(fmaxf(red[0], red[1]), fmaxf(red[2], red[3]));
  float scale = fmaxf(amax, 1e-12f) * (1.0f / 448.0f);
  xq[off] = (unsigned char)f32_to_e4m3(a / scale);
  if (i == 0) xs[(size_t)t * KB_H + kb] = scale;
}

__global__ __launch_bounds__(256)
void k_wq(const float* __restrict__ w, unsigned char* __restrict__ q, long long n) {
  long long i = ((long long)blockIdx.x * blockDim.x + threadIdx.x) * 16;
  if (i >= n) return;
  const float4* p = (const float4*)(w + i);
  uint4 o;
  o.x = pack4_e4m3(p[0]); o.y = pack4_e4m3(p[1]);
  o.z = pack4_e4m3(p[2]); o.w = pack4_e4m3(p[3]);
  *(uint4*)(q + i) = o;
}

__global__ __launch_bounds__(256)
void k_gateup(const unsigned char* __restrict__ xq, const float* __restrict__ xs,
              const unsigned char* __restrict__ w8, const float* __restrict__ gus,
              const int* __restrict__ rows, const int* __restrict__ offs,
              const int* __restrict__ te, const int* __restrict__ tm,
              const int* __restrict__ nt,
              unsigned char* __restrict__ hq, float* __restrict__ hs) {
  __shared__ __align__(16) unsigned int ldsA[2][128 * 32];
  __shared__ __align__(16) unsigned int ldsB[2][256 * 32];
  __shared__ float ldsSA[2][128];
  __shared__ float ldsSB[32];
  __shared__ int   ldsTok[128];

  const int tile = blockIdx.y;
  if (tile >= nt[0]) return;
  const int e  = te[tile];
  const int mt = tm[tile];
  const int jn = blockIdx.x;
  const int row0   = offs[e] + mt * 128;
  const int rowEnd = offs[e + 1];

  const int tid  = threadIdx.x;
  const int wave = tid >> 5;
  const int lane = tid & 31;
  const int half = lane >> 4;
  const int ln   = lane & 15;

  if (tid < 128) {
    int slot = row0 + tid;
    ldsTok[tid] = (slot < rowEnd) ? rows[slot] : 0;
  }
  if (tid < 32) {
    int nb = (tid < 16) ? jn : (8 + jn);
    ldsSB[tid] = gus[(size_t)e * 256 + (size_t)nb * 16 + (tid & 15)];
  }
  __syncthreads();

  auto stage = [&](int buf, int kb) {
    {
      int r = tid >> 1, h64 = tid & 1;
      const unsigned char* g = xq + (size_t)ldsTok[r] * H_DIM +
                               (size_t)kb * 128 + h64 * 64;
      unsigned int* l = &ldsA[buf][r * 32 + h64 * 16];
      #pragma unroll
      for (int i = 0; i < 4; ++i) async_cp16(g + i * 16, l + i * 4);
    }
    {
      int grow = jn * 128 + (tid & 127) + (tid >> 7) * I_DIM;
      const unsigned char* g = w8 + ((size_t)e * 2 * I_DIM + grow) * H_DIM +
                               (size_t)kb * 128;
      unsigned int* l = &ldsB[buf][tid * 32];
      #pragma unroll
      for (int i = 0; i < 8; ++i) async_cp16(g + i * 16, l + i * 4);
    }
    if (tid < 128) ldsSA[buf][tid] = xs[(size_t)ldsTok[tid] * KB_H + kb];
  };

  float accg[8][8];
  float accu[8][8];
  #pragma unroll
  for (int st = 0; st < 8; ++st)
    #pragma unroll
    for (int v = 0; v < 8; ++v) { accg[st][v] = 0.f; accu[st][v] = 0.f; }

  v8f cz;
  #pragma unroll
  for (int i = 0; i < 8; ++i) cz[i] = 0.f;

  stage(0, 0);

  for (int kb = 0; kb < KB_H; ++kb) {
    const int cur = kb & 1;
    if (kb + 1 < KB_H) {
      stage(cur ^ 1, kb + 1);
      ASYNC_WAIT(12);
    } else {
      ASYNC_WAIT(0);
    }
    if (kb + 2 < KB_H) {
      int grow = jn * 128 + (tid & 127) + (tid >> 7) * I_DIM;
      __builtin_prefetch(w8 + ((size_t)e * 2 * I_DIM + grow) * H_DIM +
                         (size_t)(kb + 2) * 128, 0, 0);
    }
    __syncthreads();

    const unsigned int* pA  = ldsA[cur];
    const unsigned int* pB  = ldsB[cur];
    const float*        pSA = ldsSA[cur];

    v16i a;
    const int arow = wave * 16 + ln;
    #pragma unroll
    for (int v = 0; v < 16; ++v)
      a[v] = (int)pA[arow * 32 + (v >> 2) * 8 + (v & 3) + half * 4];

    float cg[8], cu[8];
    {
      const float sbg = ldsSB[kb];
      const float sbu = ldsSB[16 + kb];
      #pragma unroll
      for (int v = 0; v < 8; ++v) {
        float sa = pSA[wave * 16 + v + half * 8];
        cg[v] = sa * sbg;
        cu[v] = sa * sbu;
      }
    }

    #pragma unroll
    for (int st = 0; st < 8; ++st) {
      v16i b;
      {
        const unsigned int* bp = &pB[(st * 16 + ln) * 32 + half * 4];
        #pragma unroll
        for (int q4 = 0; q4 < 4; ++q4) {
          const uint4 t4 = *(const uint4*)(bp + q4 * 8);
          b[q4*4+0]=(int)t4.x; b[q4*4+1]=(int)t4.y;
          b[q4*4+2]=(int)t4.z; b[q4*4+3]=(int)t4.w;
        }
      }
      v8f p = __builtin_amdgcn_wmma_f32_16x16x128_fp8_fp8(a, b, (short)0, cz,
                                                          false, false);
      #pragma unroll
      for (int v = 0; v < 8; ++v) accg[st][v] += cg[v] * p[v];
      {
        const unsigned int* bp = &pB[(128 + st * 16 + ln) * 32 + half * 4];
        #pragma unroll
        for (int q4 = 0; q4 < 4; ++q4) {
          const uint4 t4 = *(const uint4*)(bp + q4 * 8);
          b[q4*4+0]=(int)t4.x; b[q4*4+1]=(int)t4.y;
          b[q4*4+2]=(int)t4.z; b[q4*4+3]=(int)t4.w;
        }
      }
      v8f p2 = __builtin_amdgcn_wmma_f32_16x16x128_fp8_fp8(a, b, (short)0, cz,
                                                           false, false);
      #pragma unroll
      for (int v = 0; v < 8; ++v) accu[st][v] += cu[v] * p2[v];
    }
    __syncthreads();
  }

  #pragma unroll
  for (int v = 0; v < 8; ++v) {
    float gated[8];
    float am = 0.f;
    #pragma unroll
    for (int st = 0; st < 8; ++st) {
      float g = accg[st][v], u = accu[st][v];
      float hval = (g / (1.f + __expf(-g))) * u;
      gated[st] = hval;
      am = fmaxf(am, fabsf(hval));
    }
    #pragma unroll
    for (int m = 1; m < 16; m <<= 1) am = fmaxf(am, __shfl_xor(am, m, 32));
    float sc  = fmaxf(am, 1e-12f) * (1.0f / 448.0f);
    float inv = 448.0f / fmaxf(am, 1e-12f);
    int rowl = wave * 16 + v + 8 * half;
    int slot = row0 + rowl;
    if (slot < rowEnd) {
      if (ln == 0) hs[(size_t)slot * KB_I + jn] = sc;
      #pragma unroll
      for (int st = 0; st < 8; ++st)
        hq[(size_t)slot * I_DIM + jn * 128 + st * 16 + ln] =
            (unsigned char)f32_to_e4m3(gated[st] * inv);
    }
  }
}

__global__ __launch_bounds__(256)
void k_down(const unsigned char* __restrict__ hq, const float* __restrict__ hs,
            const unsigned char* __restrict__ w8, const float* __restrict__ dsc,
            const int* __restrict__ rows, const float* __restrict__ roww,
            const int* __restrict__ offs, const int* __restrict__ te,
            const int* __restrict__ tm, const int* __restrict__ nt,
            float* __restrict__ out) {
  __shared__ __align__(16) unsigned int ldsA[2][128 * 32];
  __shared__ __align__(16) unsigned int ldsB[2][64 * 32];
  __shared__ float ldsSA[2][128];
  __shared__ float ldsSB[8];
  __shared__ int   ldsTok[128];
  __shared__ float ldsCW[128];

  const int tile = blockIdx.y;
  if (tile >= nt[0]) return;
  const int e  = te[tile];
  const int mt = tm[tile];
  const int n0 = blockIdx.x * 64;
  const int nb = blockIdx.x >> 1;
  const int row0   = offs[e] + mt * 128;
  const int rowEnd = offs[e + 1];

  const int tid  = threadIdx.x;
  const int wave = tid >> 5;
  const int lane = tid & 31;
  const int half = lane >> 4;
  const int ln   = lane & 15;

  if (tid < 128) {
    int slot = row0 + tid;
    bool ok = slot < rowEnd;
    ldsTok[tid] = ok ? rows[slot] : 0;
    ldsCW[tid]  = ok ? roww[slot] : 0.f;
  }
  if (tid < 8) ldsSB[tid] = dsc[(size_t)e * 128 + (size_t)nb * 8 + tid];
  __syncthreads();

  auto stage = [&](int buf, int kb) {
    {
      int r = tid >> 1, h64 = tid & 1;
      int slot = row0 + r;
      int slotc = (slot < rowEnd) ? slot : (rowEnd - 1);
      const unsigned char* g = hq + (size_t)slotc * I_DIM +
                               (size_t)kb * 128 + h64 * 64;
      unsigned int* l = &ldsA[buf][r * 32 + h64 * 16];
      #pragma unroll
      for (int i = 0; i < 4; ++i) async_cp16(g + i * 16, l + i * 4);
    }
    {
      int c = tid >> 2, qt = tid & 3;
      const unsigned char* g = w8 + ((size_t)e * H_DIM + n0 + c) * I_DIM +
                               (size_t)kb * 128 + qt * 32;
      unsigned int* l = &ldsB[buf][c * 32 + qt * 8];
      #pragma unroll
      for (int i = 0; i < 2; ++i) async_cp16(g + i * 16, l + i * 4);
    }
    if (tid < 128)
      ldsSA[buf][tid] = (row0 + tid < rowEnd)
                            ? hs[(size_t)(row0 + tid) * KB_I + kb] : 0.f;
  };

  float acc[4][8];
  #pragma unroll
  for (int st = 0; st < 4; ++st)
    #pragma unroll
    for (int v = 0; v < 8; ++v) acc[st][v] = 0.f;

  v8f cz;
  #pragma unroll
  for (int i = 0; i < 8; ++i) cz[i] = 0.f;

  stage(0, 0);

  for (int kb = 0; kb < KB_I; ++kb) {
    const int cur = kb & 1;
    if (kb + 1 < KB_I) {
      stage(cur ^ 1, kb + 1);
      ASYNC_WAIT(6);
    } else {
      ASYNC_WAIT(0);
    }
    __syncthreads();

    const unsigned int* pA  = ldsA[cur];
    const unsigned int* pB  = ldsB[cur];
    const float*        pSA = ldsSA[cur];

    v16i a;
    const int arow = wave * 16 + ln;
    #pragma unroll
    for (int v = 0; v < 16; ++v)
      a[v] = (int)pA[arow * 32 + (v >> 2) * 8 + (v & 3) + half * 4];
    float cs[8];
    {
      const float sb = ldsSB[kb];
      #pragma unroll
      for (int v = 0; v < 8; ++v) cs[v] = pSA[wave * 16 + v + half * 8] * sb;
    }

    #pragma unroll
    for (int st = 0; st < 4; ++st) {
      v16i b;
      const unsigned int* bp = &pB[(st * 16 + ln) * 32 + half * 4];
      #pragma unroll
      for (int q4 = 0; q4 < 4; ++q4) {
        const uint4 t4 = *(const uint4*)(bp + q4 * 8);
        b[q4*4+0]=(int)t4.x; b[q4*4+1]=(int)t4.y;
        b[q4*4+2]=(int)t4.z; b[q4*4+3]=(int)t4.w;
      }
      v8f p = __builtin_amdgcn_wmma_f32_16x16x128_fp8_fp8(a, b, (short)0, cz,
                                                          false, false);
      #pragma unroll
      for (int v = 0; v < 8; ++v) acc[st][v] += cs[v] * p[v];
    }
    __syncthreads();
  }

  #pragma unroll
  for (int v = 0; v < 8; ++v) {
    int rowl = wave * 16 + v + 8 * half;
    int slot = row0 + rowl;
    if (slot < rowEnd) {
      int   tok = ldsTok[rowl];
      float cw  = ldsCW[rowl];
      #pragma unroll
      for (int st = 0; st < 4; ++st)
        atomicAdd(out + (size_t)tok * H_DIM + n0 + st * 16 + ln,
                  cw * acc[st][v]);
    }
  }
}

extern "C" void kernel_launch(void* const* d_in, const int* in_sizes, int n_in,
                              void* d_out, int out_size, void* d_ws, size_t ws_size,
                              hipStream_t stream) {
  const float* x   = (const float*)d_in[0];
  const float* wgu = (const float*)d_in[1];
  const float* sgu = (const float*)d_in[2];
  const float* wd  = (const float*)d_in[3];
  const float* sdn = (const float*)d_in[4];
  const float* tkw = (const float*)d_in[5];
  const int*   tki = (const int*)d_in[6];
  float* out = (float*)d_out;
  char*  ws  = (char*)d_ws;

  int*   counts = (int*)(ws + OFF_COUNTS);
  int*   fill   = (int*)(ws + OFF_FILL);
  int*   offs   = (int*)(ws + OFF_OFFS);
  int*   nt     = (int*)(ws + OFF_NT);
  int*   te     = (int*)(ws + OFF_TE);
  int*   tm     = (int*)(ws + OFF_TM);
  float* xs     = (float*)(ws + OFF_XS);
  unsigned char* xq = (unsigned char*)(ws + OFF_XQ);
  int*   rowsA  = (int*)(ws + OFF_ROWS);
  float* rowwA  = (float*)(ws + OFF_ROWW);
  float* hs     = (float*)(ws + OFF_HS);
  unsigned char* hq   = (unsigned char*)(ws + OFF_HQ);
  unsigned char* w8gu = (unsigned char*)(ws + OFF_W8GU);
  unsigned char* w8d  = (unsigned char*)(ws + OFF_W8D);

  k_zero_small<<<1, 64, 0, stream>>>(counts, fill);
  k_zero4<<<(T_TOK * H_DIM / 4 + 255) / 256, 256, 0, stream>>>(
      out, (long long)T_TOK * H_DIM / 4);
  k_actq<<<T_TOK * KB_H, 128, 0, stream>>>(x, xq, xs);
  k_wq<<<(int)((GU_ELEMS / 16 + 255) / 256), 256, 0, stream>>>(wgu, w8gu, GU_ELEMS);
  k_wq<<<(int)((D_ELEMS  / 16 + 255) / 256), 256, 0, stream>>>(wd, w8d, D_ELEMS);
  k_count<<<NPAIR / 256, 256, 0, stream>>>(tki, counts);
  k_scan<<<1, 32, 0, stream>>>(counts, offs, te, tm, nt);
  k_fill<<<NPAIR / 256, 256, 0, stream>>>(tki, tkw, offs, fill, rowsA, rowwA);
  k_gateup<<<dim3(8, MAXTILE), 256, 0, stream>>>(xq, xs, w8gu, sgu, rowsA, offs,
                                                 te, tm, nt, hq, hs);
  k_down<<<dim3(32, MAXTILE), 256, 0, stream>>>(hq, hs, w8d, sdn, rowsA, rowwA,
                                                offs, te, tm, nt, out);
}
